// JanusAttention_50440095924322
// MI455X (gfx1250) — matrix operand order repaired and checked by reading, not compiled, not run on hardware
//
#include <hip/hip_runtime.h>
#include <hip/hip_bf16.h>

#define BB 4
#define TT 1024
#define EE 1024
#define HH 16
#define DD 64

typedef __attribute__((ext_vector_type(16))) __bf16 bf16x16;
typedef __attribute__((ext_vector_type(8)))  float  f32x8;

#if defined(__AMDGCN__) && __has_builtin(__builtin_amdgcn_tensor_load_to_lds)
#define HAVE_TDM 1
#else
#define HAVE_TDM 0
#endif

#if HAVE_TDM
typedef unsigned int u32x4 __attribute__((ext_vector_type(4)));
typedef int          i32x8 __attribute__((ext_vector_type(8)));
typedef int          i32x4 __attribute__((ext_vector_type(4)));

// byte offset of a __shared__ object inside the wave's LDS allocation
__device__ static inline unsigned lds_off_of(const void* p) {
  return (unsigned)(unsigned long long)
      (const __attribute__((address_space(3))) char*)(const char*)p;
}

// Issue TENSOR_LOAD_TO_LDS for a 2-D bf16 tile:
//   tile_w elems per row, tile_h rows, row stride = stride_elems (global),
//   LDS rows padded: after every (1<<pi)*8 bytes insert (pa+1)*4 bytes.
__device__ static inline void tdm_load_2d(unsigned lds_byte_off, const void* gptr,
                                          unsigned tile_w, unsigned tile_h,
                                          unsigned stride_elems,
                                          unsigned pi, unsigned pa) {
  unsigned long long ga = (unsigned long long)gptr;
  u32x4 g0;
  g0[0] = 1u;                                     // count=1, user descriptor
  g0[1] = lds_byte_off;                           // lds_addr
  g0[2] = (unsigned)(ga & 0xffffffffu);           // global_addr[31:0]
  g0[3] = (unsigned)((ga >> 32) & 0x1ffffffu) | (2u << 30);  // addr[56:32] | type=2
  i32x8 g1;
  g1[0] = (int)((1u << 16)        // data_size = 2 bytes
              | (1u << 20)        // pad_enable
              | (pi << 22)        // pad_interval code
              | (pa << 25));      // pad_amount code
  g1[1] = (int)(tile_w << 16);                    // tensor_dim0[15:0]   (== tile_w)
  g1[2] = (int)((tile_w >> 16) | (tile_h << 16)); // tensor_dim0 hi | tensor_dim1 lo
  g1[3] = (int)((tile_h >> 16) | (tile_w << 16)); // tensor_dim1 hi | tile_dim0
  g1[4] = (int)tile_h;                            // tile_dim1 | tile_dim2=0
  g1[5] = (int)stride_elems;                      // tensor_dim0_stride[31:0]
  g1[6] = 0;
  g1[7] = 0;
  i32x4 z4 = {0, 0, 0, 0};
  i32x8 z8 = {0, 0, 0, 0, 0, 0, 0, 0};
  __builtin_amdgcn_tensor_load_to_lds(g0, g1, z4, z4, z8, 0);
}
#endif

// ---------------------------------------------------------------------------
// WMMA fragment helpers (A/B use contiguous k = 16*half + element, wave32)
// ---------------------------------------------------------------------------
__device__ inline bf16x16 load_a_frag(const __bf16* p, int ld) {
  int lane = threadIdx.x & 31;
  int m = lane & 15, half = lane >> 4;
  const __bf16* row = p + m * ld;
  bf16x16 a;
#pragma unroll
  for (int v = 0; v < 8; ++v) {
    int kb = half * 16 + 2 * v;
    a[2 * v]     = row[kb];
    a[2 * v + 1] = row[kb + 1];
  }
  return a;
}

__device__ inline bf16x16 load_b_frag_nk(const __bf16* p, int ld) {
  int lane = threadIdx.x & 31;
  int n = lane & 15, half = lane >> 4;
  const __bf16* row = p + n * ld;
  bf16x16 b;
#pragma unroll
  for (int v = 0; v < 8; ++v) {
    int k = half * 16 + 2 * v;
    b[2 * v]     = row[k];
    b[2 * v + 1] = row[k + 1];
  }
  return b;
}

__device__ inline bf16x16 load_b_frag_kn(const __bf16* p, int ld) {
  int lane = threadIdx.x & 31;
  int n = lane & 15, half = lane >> 4;
  bf16x16 b;
#pragma unroll
  for (int v = 0; v < 8; ++v) {
    int k = half * 16 + 2 * v;
    b[2 * v]     = p[k * ld + n];
    b[2 * v + 1] = p[(k + 1) * ld + n];
  }
  return b;
}

__device__ inline f32x8 wmma_bf16(bf16x16 a, bf16x16 b, f32x8 c) {
  return __builtin_amdgcn_wmma_f32_16x16x32_bf16(false, a, false, b,
                                                 (short)0, c, false, false);
}

// ---------------------------------------------------------------------------
// Conversion kernels
// ---------------------------------------------------------------------------
__global__ void cvt_bf16(const float* __restrict__ in, __bf16* __restrict__ out, int n) {
  for (int i = blockIdx.x * blockDim.x + threadIdx.x; i < n; i += gridDim.x * blockDim.x)
    out[i] = (__bf16)in[i];
}

__global__ void transpose_bf16(const float* __restrict__ in, __bf16* __restrict__ out, int N) {
  int i = blockIdx.x * blockDim.x + threadIdx.x;
  if (i < N * N) {
    int e = i / N, f = i % N;
    out[i] = (__bf16)in[f * N + e];
  }
}

// ---------------------------------------------------------------------------
// bf16 GEMM: C[M,N] = A[M,K] @ B[N,K]^T, 128x128 tile, 8 waves,
// TDM double-buffered global->LDS tile DMA when available.
// ---------------------------------------------------------------------------
__global__ __launch_bounds__(256) void gemm_bf16_nt(
    const __bf16* __restrict__ A, const __bf16* __restrict__ Bm,
    __bf16* __restrict__ outb, float* __restrict__ outf,
    int M, int N, int K) {
  __shared__ __bf16 sA[2][128 * 40];
  __shared__ __bf16 sB[2][128 * 40];
  int tid  = threadIdx.x;
  int wave = tid >> 5;
  int m0 = blockIdx.y * 128;
  int n0 = blockIdx.x * 128;
  int wm = (wave & 3) * 32;
  int wn = (wave >> 2) * 64;

  f32x8 acc[2][4] = {};
  int nk = K / 32;

#if HAVE_TDM
  if (wave == 0) {
    tdm_load_2d(lds_off_of(sA[0]), A  + (size_t)m0 * K, 32, 128, K, 3, 3);
    tdm_load_2d(lds_off_of(sB[0]), Bm + (size_t)n0 * K, 32, 128, K, 3, 3);
  }
#endif

  for (int kt = 0; kt < nk; ++kt) {
    __syncthreads();   // all waves done with buffer being overwritten next
#if HAVE_TDM
    if (wave == 0) {
      if (kt + 1 < nk) {
        tdm_load_2d(lds_off_of(sA[(kt + 1) & 1]),
                    A + (size_t)m0 * K + (kt + 1) * 32, 32, 128, K, 3, 3);
        tdm_load_2d(lds_off_of(sB[(kt + 1) & 1]),
                    Bm + (size_t)n0 * K + (kt + 1) * 32, 32, 128, K, 3, 3);
        __builtin_amdgcn_s_wait_tensorcnt((short)2);  // current tile pair done
      } else {
        __builtin_amdgcn_s_wait_tensorcnt((short)0);
      }
    }
#else
    {
      int k0 = kt * 32;
#pragma unroll
      for (int p = 0; p < 2; ++p) {
        int idx = tid + p * 256;
        int row = idx >> 2, cg = idx & 3;
        const uint4* ga = (const uint4*)(A  + (size_t)(m0 + row) * K + k0);
        const uint4* gb = (const uint4*)(Bm + (size_t)(n0 + row) * K + k0);
        *(uint4*)(&sA[kt & 1][row * 40 + cg * 8]) = ga[cg];
        *(uint4*)(&sB[kt & 1][row * 40 + cg * 8]) = gb[cg];
      }
    }
#endif
    __syncthreads();
    const __bf16* cA = sA[kt & 1];
    const __bf16* cB = sB[kt & 1];
    bf16x16 aF[2], bF[4];
#pragma unroll
    for (int s = 0; s < 2; ++s) aF[s] = load_a_frag(&cA[(wm + s * 16) * 40], 40);
#pragma unroll
    for (int t = 0; t < 4; ++t) bF[t] = load_b_frag_nk(&cB[(wn + t * 16) * 40], 40);
#pragma unroll
    for (int s = 0; s < 2; ++s)
#pragma unroll
      for (int t = 0; t < 4; ++t) acc[s][t] = wmma_bf16(aF[s], bF[t], acc[s][t]);
  }

  int lane = tid & 31, n = lane & 15, half = lane >> 4;
  if (outf) {
#pragma unroll
    for (int s = 0; s < 2; ++s)
#pragma unroll
      for (int t = 0; t < 4; ++t)
#pragma unroll
        for (int r = 0; r < 8; ++r)
          outf[(size_t)(m0 + wm + s * 16 + r + half * 8) * N + n0 + wn + t * 16 + n] =
              acc[s][t][r];
  } else {
#pragma unroll
    for (int s = 0; s < 2; ++s)
#pragma unroll
      for (int t = 0; t < 4; ++t)
#pragma unroll
        for (int r = 0; r < 8; ++r)
          outb[(size_t)(m0 + wm + s * 16 + r + half * 8) * N + n0 + wn + t * 16 + n] =
              (__bf16)acc[s][t][r];
  }
}

// ---------------------------------------------------------------------------
// s2[b,h,t] = 0.125 * dot(x[b,t,:], wr[h,:,t])
// Block = (b, 256 t's); lane->t so wr reads are coalesced on its fast axis;
// x staged via LDS transpose; 16 per-head accumulators in registers.
// ---------------------------------------------------------------------------
__global__ __launch_bounds__(256) void s2_kernel(const float* __restrict__ x,
                                                 const float* __restrict__ wr,
                                                 float* __restrict__ s2) {
  __shared__ float sX[32][257];
  int tid = threadIdx.x;
  int t0 = (blockIdx.x % (TT / 256)) * 256;
  int b  = blockIdx.x / (TT / 256);
  float acc[HH];
#pragma unroll
  for (int h = 0; h < HH; ++h) acc[h] = 0.f;

  for (int e0 = 0; e0 < EE; e0 += 32) {
    __syncthreads();
#pragma unroll
    for (int p = 0; p < 8; ++p) {
      int idx = tid + p * 256;          // 0..2047
      int row = idx >> 3, cg = idx & 7; // 256 rows x 8 float4
      float4 v = *(const float4*)(x + (size_t)(b * TT + t0 + row) * EE + e0 + cg * 4);
      sX[cg * 4 + 0][row] = v.x;
      sX[cg * 4 + 1][row] = v.y;
      sX[cg * 4 + 2][row] = v.z;
      sX[cg * 4 + 3][row] = v.w;
    }
    __syncthreads();
    for (int e = 0; e < 32; ++e) {
      float xv = sX[e][tid];
      int ge = e0 + e;
#pragma unroll
      for (int h = 0; h < HH; ++h)
        acc[h] += xv * wr[((size_t)h * EE + ge) * TT + t0 + tid];
    }
  }
#pragma unroll
  for (int h = 0; h < HH; ++h)
    s2[(b * HH + h) * TT + t0 + tid] = acc[h] * 0.125f;
}

// ---------------------------------------------------------------------------
// sc[b,t] = (1/32) * dot(x[b,t,:], echo_back[b,t,:])
// ---------------------------------------------------------------------------
__global__ __launch_bounds__(256) void sc_kernel(const float* __restrict__ x,
                                                 const __bf16* __restrict__ eb,
                                                 float* __restrict__ sc) {
  int gid  = blockIdx.x * 8 + (threadIdx.x >> 5);
  int lane = threadIdx.x & 31;
  float sum = 0.f;
  for (int e = lane; e < EE; e += 32)
    sum += x[(size_t)gid * EE + e] * (float)eb[(size_t)gid * EE + e];
#pragma unroll
  for (int d = 16; d; d >>= 1) sum += __shfl_xor(sum, d, 32);
  if (lane == 0) sc[gid] = sum * 0.03125f;
}

// ---------------------------------------------------------------------------
// Branch 2 online-softmax prefix scan, one 64-thread block per (b,h)
// ---------------------------------------------------------------------------
__global__ __launch_bounds__(64) void scan2_kernel(const float* __restrict__ s2,
                                                   const __bf16* __restrict__ rv,
                                                   float* __restrict__ out2) {
  int h = blockIdx.x % HH, b = blockIdx.x / HH;
  int d = threadIdx.x;  // 0..63
  float m = -3e38f, Z = 0.f, acc = 0.f;
  for (int j = 0; j < TT; ++j) {
    if (j + 16 < TT)
      __builtin_prefetch(rv + (size_t)(b * TT + j + 16) * EE + h * DD + d, 0, 1);
    float s  = s2[(b * HH + h) * TT + j];
    float mn = fmaxf(m, s);
    float rf = __expf(m - mn);
    float w  = __expf(s - mn);
    float v  = (float)rv[(size_t)(b * TT + j) * EE + h * DD + d];
    Z   = Z * rf + w;
    acc = acc * rf + w * v;
    out2[((size_t)(b * HH + h) * TT + j) * DD + d] = acc / Z;
    m = mn;
  }
}

// ---------------------------------------------------------------------------
// Fused flash kernel: branch 1 + branch 3 with online softmax, TDM tile DMA,
// gated combine with branch 2. Grid (T/128, H, B), 8 waves.
// ---------------------------------------------------------------------------
__global__ __launch_bounds__(256) void flash3_kernel(
    const __bf16* __restrict__ Qb, const __bf16* __restrict__ Kb,
    const __bf16* __restrict__ Vb, const __bf16* __restrict__ Eb,
    const float* __restrict__ sc,  const float* __restrict__ out2,
    const float* __restrict__ gate, __bf16* __restrict__ comb) {
  __shared__ __bf16 sQ[128 * 72];
  __shared__ __bf16 sK[32 * 72];
  __shared__ __bf16 sV[32 * 72];
  __shared__ __bf16 sJ[32 * 72];
  __shared__ float  sS[32];
  __shared__ __bf16 sP[8][16 * 40];

  int tid = threadIdx.x, lane = tid & 31, wave = tid >> 5;
  int n = lane & 15, half = lane >> 4;
  int i0 = blockIdx.x * 128;
  int h  = blockIdx.y, b = blockIdx.z;
  const float scale = 0.125f;

#if HAVE_TDM
  if (wave == 0) {
    tdm_load_2d(lds_off_of(sQ), Qb + (size_t)(b * TT + i0) * EE + h * DD,
                64, 128, EE, 4, 3);
    __builtin_amdgcn_s_wait_tensorcnt((short)0);
  }
#else
#pragma unroll
  for (int p = 0; p < 2; ++p) {
    int idx = tid + p * 256;
    int row = idx >> 3, cg = idx & 7;
    const uint4* g = (const uint4*)(Qb + (size_t)(b * TT + i0 + row) * EE + h * DD);
    *(uint4*)(&sQ[row * 72 + cg * 8]) = g[cg];
  }
#endif
  __syncthreads();

  int q0 = i0 + wave * 16;
  bf16x16 qF0 = load_a_frag(&sQ[(wave * 16) * 72], 72);
  bf16x16 qF1 = load_a_frag(&sQ[(wave * 16) * 72 + 32], 72);

  float sci[8];
#pragma unroll
  for (int r = 0; r < 8; ++r) sci[r] = sc[b * TT + q0 + r + half * 8];

  f32x8 o1[4] = {};
  f32x8 o3[4] = {};
  float m1[8], l1[8], m3[8], l3[8];
#pragma unroll
  for (int r = 0; r < 8; ++r) { m1[r] = -3e38f; l1[r] = 0.f; m3[r] = -3e38f; l3[r] = 0.f; }

  int njt = (i0 + 128) / 32;
  for (int jt = 0; jt < njt; ++jt) {
    int j0 = jt * 32;
    __syncthreads();
#if HAVE_TDM
    if (wave == 0) {
      tdm_load_2d(lds_off_of(sK), Kb + (size_t)(b * TT + j0) * EE + h * DD, 64, 32, EE, 4, 3);
      tdm_load_2d(lds_off_of(sV), Vb + (size_t)(b * TT + j0) * EE + h * DD, 64, 32, EE, 4, 3);
      tdm_load_2d(lds_off_of(sJ), Eb + (size_t)(b * TT + j0) * EE + h * DD, 64, 32, EE, 4, 3);
      __builtin_amdgcn_s_wait_tensorcnt((short)0);
    }
    if (tid < 32) sS[tid] = sc[b * TT + j0 + tid];
#else
    {
      int row = tid >> 3, cg = tid & 7;
      const uint4* gk = (const uint4*)(Kb + (size_t)(b * TT + j0 + row) * EE + h * DD);
      const uint4* gv = (const uint4*)(Vb + (size_t)(b * TT + j0 + row) * EE + h * DD);
      const uint4* gj = (const uint4*)(Eb + (size_t)(b * TT + j0 + row) * EE + h * DD);
      *(uint4*)(&sK[row * 72 + cg * 8]) = gk[cg];
      *(uint4*)(&sV[row * 72 + cg * 8]) = gv[cg];
      *(uint4*)(&sJ[row * 72 + cg * 8]) = gj[cg];
      if (tid < 32) sS[tid] = sc[b * TT + j0 + tid];
    }
#endif
    __syncthreads();
    if (j0 > q0 + 15) continue;

    // ---- branch 1: q . k^T via WMMA ----
    f32x8 c1[2] = {};
#pragma unroll
    for (int js = 0; js < 2; ++js) {
      bf16x16 b0 = load_b_frag_nk(&sK[(js * 16) * 72], 72);
      bf16x16 b1 = load_b_frag_nk(&sK[(js * 16) * 72 + 32], 72);
      c1[js] = wmma_bf16(qF0, b0, c1[js]);
      c1[js] = wmma_bf16(qF1, b1, c1[js]);
    }

#pragma unroll
    for (int r = 0; r < 8; ++r) {
      int q   = q0 + r + half * 8;
      int k0i = j0 + n, k1i = j0 + 16 + n;
      float s0 = (k0i > q) ? -3e38f : c1[0][r] * scale;
      float s1 = (k1i > q) ? -3e38f : c1[1][r] * scale;
      float mx = fmaxf(s0, s1);
#pragma unroll
      for (int d = 1; d < 16; d <<= 1) mx = fmaxf(mx, __shfl_xor(mx, d, 32));
      float mn  = fmaxf(m1[r], mx);
      float fac = __expf(m1[r] - mn);
      m1[r] = mn;
      float e0 = __expf(s0 - mn), e1 = __expf(s1 - mn);
      float rs = e0 + e1;
#pragma unroll
      for (int d = 1; d < 16; d <<= 1) rs += __shfl_xor(rs, d, 32);
      l1[r] = l1[r] * fac + rs;
#pragma unroll
      for (int t = 0; t < 4; ++t) o1[t][r] *= fac;
      sP[wave][(r + half * 8) * 40 + n]      = (__bf16)e0;
      sP[wave][(r + half * 8) * 40 + 16 + n] = (__bf16)e1;
    }
    {
      bf16x16 pF = load_a_frag(&sP[wave][0], 40);   // DS in-order per wave
#pragma unroll
      for (int t = 0; t < 4; ++t) {
        bf16x16 vF = load_b_frag_kn(&sV[t * 16], 72);
        o1[t] = wmma_bf16(pF, vF, o1[t]);
      }
    }

    // ---- branch 3: rank-1 scores sc_i * sc_j ----
#pragma unroll
    for (int r = 0; r < 8; ++r) {
      int q   = q0 + r + half * 8;
      int k0i = j0 + n, k1i = j0 + 16 + n;
      float s0 = (k0i > q) ? -3e38f : sci[r] * sS[n];
      float s1 = (k1i > q) ? -3e38f : sci[r] * sS[16 + n];
      float mx = fmaxf(s0, s1);
#pragma unroll
      for (int d = 1; d < 16; d <<= 1) mx = fmaxf(mx, __shfl_xor(mx, d, 32));
      float mn  = fmaxf(m3[r], mx);
      float fac = __expf(m3[r] - mn);
      m3[r] = mn;
      float e0 = __expf(s0 - mn), e1 = __expf(s1 - mn);
      float rs = e0 + e1;
#pragma unroll
      for (int d = 1; d < 16; d <<= 1) rs += __shfl_xor(rs, d, 32);
      l3[r] = l3[r] * fac + rs;
#pragma unroll
      for (int t = 0; t < 4; ++t) o3[t][r] *= fac;
      sP[wave][(r + half * 8) * 40 + n]      = (__bf16)e0;
      sP[wave][(r + half * 8) * 40 + 16 + n] = (__bf16)e1;
    }
    {
      bf16x16 pF = load_a_frag(&sP[wave][0], 40);
#pragma unroll
      for (int t = 0; t < 4; ++t) {
        bf16x16 jF = load_b_frag_kn(&sJ[t * 16], 72);
        o3[t] = wmma_bf16(pF, jF, o3[t]);
      }
    }
  }

  float g0 = gate[h * 3 + 0], g1 = gate[h * 3 + 1], g2 = gate[h * 3 + 2];
  float gm = fmaxf(g0, fmaxf(g1, g2));
  float e0 = __expf(g0 - gm), e1 = __expf(g1 - gm), e2 = __expf(g2 - gm);
  float gs = e0 + e1 + e2;
  g0 = e0 / gs; g1 = e1 / gs; g2 = e2 / gs;

#pragma unroll
  for (int r = 0; r < 8; ++r) {
    int q = q0 + r + half * 8;
    float inv1 = 1.f / l1[r], inv3 = 1.f / l3[r];
#pragma unroll
    for (int t = 0; t < 4; ++t) {
      int d = t * 16 + n;
      float v2 = out2[((size_t)(b * HH + h) * TT + q) * DD + d];
      float v  = g0 * o1[t][r] * inv1 + g1 * v2 + g2 * o3[t][r] * inv3;
      comb[(size_t)(b * TT + q) * EE + h * DD + d] = (__bf16)v;
    }
  }
}

// ---------------------------------------------------------------------------
// Host launch
// ---------------------------------------------------------------------------
extern "C" void kernel_launch(void* const* d_in, const int* in_sizes, int n_in,
                              void* d_out, int out_size, void* d_ws, size_t ws_size,
                              hipStream_t stream) {
  const float* x    = (const float*)d_in[0];
  const float* wq   = (const float*)d_in[1];
  const float* wk   = (const float*)d_in[2];
  const float* wv   = (const float*)d_in[3];
  const float* wvr  = (const float*)d_in[4];
  const float* wr   = (const float*)d_in[5];
  const float* wj   = (const float*)d_in[6];
  const float* gate = (const float*)d_in[7];
  const float* wo   = (const float*)d_in[8];

  char* ws = (char*)d_ws;
  size_t off = 0;
  auto alloc = [&](size_t bytes) {
    char* p = ws + off;
    off = (off + bytes + 255) & ~(size_t)255;
    return p;
  };

  const size_t MT = (size_t)BB * TT;
  __bf16* xb   = (__bf16*)alloc(MT * EE * 2);
  __bf16* wqb  = (__bf16*)alloc((size_t)EE * EE * 2);
  __bf16* wkb  = (__bf16*)alloc((size_t)EE * EE * 2);
  __bf16* wvb  = (__bf16*)alloc((size_t)EE * EE * 2);
  __bf16* wvrb = (__bf16*)alloc((size_t)EE * EE * 2);
  __bf16* wjb  = (__bf16*)alloc((size_t)EE * EE * 2);
  __bf16* wjtb = (__bf16*)alloc((size_t)EE * EE * 2);
  __bf16* wob  = (__bf16*)alloc((size_t)EE * EE * 2);
  __bf16* Qb   = (__bf16*)alloc(MT * EE * 2);
  __bf16* Kb   = (__bf16*)alloc(MT * EE * 2);
  __bf16* Vb   = (__bf16*)alloc(MT * EE * 2);
  __bf16* RVb  = (__bf16*)alloc(MT * EE * 2);
  __bf16* Echo = (__bf16*)alloc(MT * EE * 2);
  __bf16* Ebk  = (__bf16*)alloc(MT * EE * 2);
  float*  scv  = (float*)alloc(MT * 4);
  float*  s2v  = (float*)alloc((size_t)BB * HH * TT * 4);
  float*  o2v  = (float*)alloc((size_t)BB * HH * TT * DD * 4);
  __bf16* comb = (__bf16*)alloc(MT * EE * 2);

  cvt_bf16<<<1024, 256, 0, stream>>>(x,   xb,   (int)(MT * EE));
  cvt_bf16<<<512,  256, 0, stream>>>(wq,  wqb,  EE * EE);
  cvt_bf16<<<512,  256, 0, stream>>>(wk,  wkb,  EE * EE);
  cvt_bf16<<<512,  256, 0, stream>>>(wv,  wvb,  EE * EE);
  cvt_bf16<<<512,  256, 0, stream>>>(wvr, wvrb, EE * EE);
  cvt_bf16<<<512,  256, 0, stream>>>(wj,  wjb,  EE * EE);
  cvt_bf16<<<512,  256, 0, stream>>>(wo,  wob,  EE * EE);
  transpose_bf16<<<(EE * EE + 255) / 256, 256, 0, stream>>>(wj, wjtb, EE);

  dim3 gg(EE / 128, (unsigned)(MT / 128));
  gemm_bf16_nt<<<gg, 256, 0, stream>>>(xb, wqb,  Qb,   nullptr, (int)MT, EE, EE);
  gemm_bf16_nt<<<gg, 256, 0, stream>>>(xb, wkb,  Kb,   nullptr, (int)MT, EE, EE);
  gemm_bf16_nt<<<gg, 256, 0, stream>>>(xb, wvb,  Vb,   nullptr, (int)MT, EE, EE);
  gemm_bf16_nt<<<gg, 256, 0, stream>>>(xb, wvrb, RVb,  nullptr, (int)MT, EE, EE);
  gemm_bf16_nt<<<gg, 256, 0, stream>>>(xb, wjb,  Echo, nullptr, (int)MT, EE, EE);
  gemm_bf16_nt<<<gg, 256, 0, stream>>>(Echo, wjtb, Ebk, nullptr, (int)MT, EE, EE);

  s2_kernel<<<BB * (TT / 256), 256, 0, stream>>>(x, wr, s2v);
  sc_kernel<<<(int)(MT / 8), 256, 0, stream>>>(x, Ebk, scv);
  scan2_kernel<<<BB * HH, 64, 0, stream>>>(s2v, RVb, o2v);

  dim3 gf(TT / 128, HH, BB);
  flash3_kernel<<<gf, 256, 0, stream>>>(Qb, Kb, Vb, Echo, scv, o2v, gate, comb);

  gemm_bf16_nt<<<gg, 256, 0, stream>>>(comb, wob, nullptr, (float*)d_out, (int)MT, EE, EE);
}
